// Attention_1821066133651
// MI455X (gfx1250) — matrix operand order repaired and checked by reading, not compiled, not run on hardware
//
#include <hip/hip_runtime.h>

// Problem constants (match reference)
#define S_LEN 2048
#define HID   2560
#define DQ    4096   // N_HEADS * HEAD_DIM
#define DKV   1024   // N_KV * HEAD_DIM
#define HD    128
#define NH    32
#define NKV   8

typedef _Float16 v8h  __attribute__((ext_vector_type(8)));
typedef _Float16 v16h __attribute__((ext_vector_type(16)));
typedef float    v8f  __attribute__((ext_vector_type(8)));

__device__ __forceinline__ v8f wmma16x16x32(v16h a, v16h b, v8f c) {
  // (neg_a, A, neg_b, B, c_mod, C, reuse_a, reuse_b)
  return __builtin_amdgcn_wmma_f32_16x16x32_f16(false, a, false, b, (short)0, c,
                                                false, false);
}

// A-fragment (16x32 f16, M x K), contiguous K order (matches load_fragB):
// lanes 0-15 : M=lane,   V0-7 = K 0..15
// lanes 16-31: M=lane-16,V0-7 = K 16..31
__device__ __forceinline__ v16h load_fragA(const _Float16* A, int lda, int m0,
                                           int k0, int lane) {
  int m = m0 + (lane & 15);
  int g = lane >> 4;
  const _Float16* p = A + (size_t)m * lda + k0 + g * 16;
  v8h lo = *(const v8h*)p;
  v8h hi = *(const v8h*)(p + 8);
  return __builtin_shufflevector(lo, hi, 0, 1, 2, 3, 4, 5, 6, 7,
                                 8, 9, 10, 11, 12, 13, 14, 15);
}

// B-fragment (32x16 f16, K x N) from an N-major buffer (ldb elems between N
// rows, K contiguous): lanes 0-15: N=lane, K 0..15 in V0..7; lanes 16-31:
// K 16..31.
__device__ __forceinline__ v16h load_fragB(const _Float16* B, int ldb, int n0,
                                           int k0, int lane) {
  int n = n0 + (lane & 15);
  int g = lane >> 4;
  return *(const v16h*)(B + (size_t)n * ldb + k0 + g * 16);
}

// One 16-byte async global->LDS copy (per-lane addresses, ASYNCcnt-tracked).
// Generic pointers to LDS carry the LDS byte offset in their low 32 bits.
__device__ __forceinline__ void async16(void* lds_dst, const void* gsrc) {
  uint32_t l = (uint32_t)(uintptr_t)lds_dst;
  uint64_t g = (uint64_t)(uintptr_t)gsrc;
  asm volatile("global_load_async_to_lds_b128 %0, %1, off"
               :
               : "v"(l), "v"(g)
               : "memory");
}

// ---------------------------------------------------------------------------
// C[M,N] = A[M,K] * B[N,K]^T (f16 in, f32 out). 4 waves, 64x64 block tile,
// 32x32 wave tile. A/B k-slabs staged in double-buffered LDS via async copies.
__global__ __launch_bounds__(128) void gemm_nt_f16f32(
    const _Float16* __restrict__ A, const _Float16* __restrict__ B,
    float* __restrict__ C, int M, int N, int K) {
  __shared__ alignas(16) _Float16 sA[2][64 * 32];
  __shared__ alignas(16) _Float16 sB[2][64 * 32];
  int tid = threadIdx.x;
  int lane = tid & 31;
  int wave = tid >> 5;
  int m0 = blockIdx.y * 64, n0 = blockIdx.x * 64;
  int mw = (wave >> 1) * 32, nw = (wave & 1) * 32;
  (void)M;

  // Cooperative staging: 64x32 f16 tile = 256 16-byte chunks; 2 per thread.
  // chunk c -> row c>>2, cols (c&3)*8 .. +7
  auto stage = [&](int buf, int k0) {
#pragma unroll
    for (int i = 0; i < 2; ++i) {
      int c = tid + i * 128;
      int r = c >> 2, col = (c & 3) * 8;
      async16(&sA[buf][c * 8], A + (size_t)(m0 + r) * K + k0 + col);
      async16(&sB[buf][c * 8], B + (size_t)(n0 + r) * K + k0 + col);
    }
  };

  int nk = K >> 5;
  stage(0, 0);
  v8f acc[2][2] = {};
  int buf = 0;
  for (int it = 0; it < nk; ++it) {
    if (it + 1 < nk) {
      stage(buf ^ 1, (it + 1) << 5);
      asm volatile("s_wait_asynccnt 0x4" ::: "memory");  // current slab done
    } else {
      asm volatile("s_wait_asynccnt 0x0" ::: "memory");
    }
    __syncthreads();  // all waves' slab data visible in LDS
    const _Float16* cA = sA[buf];
    const _Float16* cB = sB[buf];
    v16h a0 = load_fragA(cA, 32, mw, 0, lane);
    v16h a1 = load_fragA(cA, 32, mw + 16, 0, lane);
    v16h b0 = load_fragB(cB, 32, nw, 0, lane);
    v16h b1 = load_fragB(cB, 32, nw + 16, 0, lane);
    acc[0][0] = wmma16x16x32(a0, b0, acc[0][0]);
    acc[0][1] = wmma16x16x32(a0, b1, acc[0][1]);
    acc[1][0] = wmma16x16x32(a1, b0, acc[1][0]);
    acc[1][1] = wmma16x16x32(a1, b1, acc[1][1]);
    asm volatile("s_wait_dscnt 0x0" ::: "memory");  // reads done before reuse
    __syncthreads();
    buf ^= 1;
  }

  int lg = lane >> 4, ln = lane & 15;
  for (int i = 0; i < 2; ++i)
    for (int j = 0; j < 2; ++j)
      for (int r = 0; r < 8; ++r)
        C[(size_t)(m0 + mw + i * 16 + lg * 8 + r) * N +
          (n0 + nw + j * 16 + ln)] = acc[i][j][r];
}

// ---------------------------------------------------------------------------
__global__ void cvt_f32_f16(const float* __restrict__ in,
                            _Float16* __restrict__ out, size_t n) {
  size_t i = (size_t)blockIdx.x * blockDim.x + threadIdx.x;
  if (i < n) out[i] = (_Float16)in[i];
}

// V: (S, NKV, HD) f32 -> Vt: (NKV, HD, S) f16  (keys contiguous for PV B-frags)
__global__ void v_transpose(const float* __restrict__ Vf,
                            _Float16* __restrict__ Vt) {
  size_t i = (size_t)blockIdx.x * blockDim.x + threadIdx.x;
  if (i >= (size_t)S_LEN * DKV) return;
  size_t s = i >> 10;    // / (NKV*HD)
  size_t hd = i & 1023;  // h*HD + d
  Vt[hd * S_LEN + s] = (_Float16)Vf[i];
}

// RMSNorm over HD=128 + RoPE, one wave per (s, head). d = lane + 32*j; the
// rotate-half partner d^64 is j^2 in the same lane.
__global__ __launch_bounds__(128) void rmsnorm_rope(
    const float* __restrict__ X, const float* __restrict__ cosb,
    const float* __restrict__ sinb, const float* __restrict__ w,
    _Float16* __restrict__ out, int nh) {
  int lane = threadIdx.x & 31;
  int gw = blockIdx.x * 4 + (threadIdx.x >> 5);
  int s = gw / nh, h = gw % nh;
  if (s >= S_LEN) return;
  const float* x = X + ((size_t)s * nh + h) * HD;
  float v[4];
  float ss = 0.f;
  for (int j = 0; j < 4; ++j) {
    v[j] = x[lane + 32 * j];
    ss += v[j] * v[j];
  }
  for (int msk = 1; msk < 32; msk <<= 1) ss += __shfl_xor(ss, msk, 32);
  float r = rsqrtf(ss * (1.0f / HD) + 1e-6f);
  _Float16* o = out + ((size_t)s * nh + h) * HD;
  for (int j = 0; j < 4; ++j) {
    int d = lane + 32 * j;
    float xn = v[j] * r * w[d];
    float xo = v[j ^ 2] * r * w[d ^ 64];
    float c = cosb[(size_t)s * HD + d];
    float sn = sinb[(size_t)s * HD + d];
    float rot = (j < 2) ? -xo : xo;
    o[d] = (_Float16)(xn * c + rot * sn);
  }
}

// ---------------------------------------------------------------------------
// Flash attention: one wave per (head, 16-query tile). 32 keys per iteration.
__global__ __launch_bounds__(128) void attn_fwd(
    const _Float16* __restrict__ Q,   // (S, NH, HD) f16 (post norm+rope)
    const _Float16* __restrict__ Kc,  // (S, NKV, HD) f16
    const _Float16* __restrict__ Vt,  // (NKV, HD, S) f16
    _Float16* __restrict__ Ao) {      // (S, NH, HD) f16
  __shared__ alignas(16) _Float16 pbuf[4][16 * 32];  // per-wave P staging
  int lane = threadIdx.x & 31;
  int wave = threadIdx.x >> 5;
  int gw = blockIdx.x * 4 + wave;
  int h = gw >> 7;  // 128 q-tiles per head
  int qb = gw & 127;
  int kvh = h / (NH / NKV);
  int m0 = qb * 16;
  int lg = lane >> 4, ln = lane & 15;

  v16h qa[4];
  for (int t = 0; t < 4; ++t)
    qa[t] = load_fragA(Q + (size_t)h * HD, NH * HD, m0, t * 32, lane);

  v8f o[8] = {};
  float mrow[8], lrow[8];
  for (int r = 0; r < 8; ++r) {
    mrow[r] = -1e30f;
    lrow[r] = 0.f;
  }

  const _Float16* Kb = Kc + (size_t)kvh * HD;
  const _Float16* Vb = Vt + (size_t)kvh * HD * S_LEN;
  _Float16* pl = pbuf[wave];
  const float scale = 0.088388347648318447f;  // 1/sqrt(128)

  int nIter = (m0 + 16 + 31) / 32;
  for (int it = 0; it < nIter; ++it) {
    int n0 = it * 32;
    // S = Q K^T for 32 keys (two 16-col tiles), K over head dim 128
    v8f sc[2] = {};
    for (int half = 0; half < 2; ++half)
      for (int t = 0; t < 4; ++t)
        sc[half] = wmma16x16x32(
            qa[t], load_fragB(Kb, NKV * HD, n0 + half * 16, t * 32, lane),
            sc[half]);

    // scale + causal mask (C elem (m0+lg*8+r, n0+half*16+ln))
    for (int half = 0; half < 2; ++half)
      for (int r = 0; r < 8; ++r) {
        float sv = sc[half][r] * scale;
        if (n0 + half * 16 + ln > m0 + lg * 8 + r) sv = -1e30f;
        sc[half][r] = sv;
      }

    // online softmax; row stats reduced across the 16-lane group
    float alphas[8];
    for (int r = 0; r < 8; ++r) {
      float mx = fmaxf(sc[0][r], sc[1][r]);
      for (int msk = 1; msk < 16; msk <<= 1)
        mx = fmaxf(mx, __shfl_xor(mx, msk, 32));
      float mn = fmaxf(mrow[r], mx);
      float p0 = __expf(sc[0][r] - mn);
      float p1 = __expf(sc[1][r] - mn);
      float rs = p0 + p1;
      for (int msk = 1; msk < 16; msk <<= 1) rs += __shfl_xor(rs, msk, 32);
      float al = __expf(mrow[r] - mn);
      lrow[r] = lrow[r] * al + rs;
      mrow[r] = mn;
      alphas[r] = al;
      int ri = lg * 8 + r;
      pl[ri * 32 + ln] = (_Float16)p0;
      pl[ri * 32 + 16 + ln] = (_Float16)p1;
    }
    for (int j = 0; j < 8; ++j)
      for (int r = 0; r < 8; ++r) o[j][r] *= alphas[r];

    // Same-wave LDS store -> A-fragment reload; wait for store completion.
    asm volatile("s_wait_dscnt 0x0" ::: "memory");
    v16h pa = load_fragA(pl, 32, 0, 0, lane);
    // O += P @ V : 8 dim tiles of 16, K=32 keys
    for (int j = 0; j < 8; ++j)
      o[j] = wmma16x16x32(pa, load_fragB(Vb, S_LEN, j * 16, n0, lane), o[j]);
  }

  for (int j = 0; j < 8; ++j)
    for (int r = 0; r < 8; ++r) {
      int mm = m0 + lg * 8 + r;
      Ao[((size_t)mm * NH + h) * HD + j * 16 + ln] =
          (_Float16)(o[j][r] / lrow[r]);
    }
}

// ---------------------------------------------------------------------------
extern "C" void kernel_launch(void* const* d_in, const int* in_sizes, int n_in,
                              void* d_out, int out_size, void* d_ws,
                              size_t ws_size, hipStream_t stream) {
  const float* X = (const float*)d_in[0];     // (1,S,H)
  const float* cosb = (const float*)d_in[1];  // (1,S,128)
  const float* sinb = (const float*)d_in[2];
  const float* Wq = (const float*)d_in[3];    // (4096,2560)
  const float* Wk = (const float*)d_in[4];    // (1024,2560)
  const float* Wv = (const float*)d_in[5];
  const float* Wo = (const float*)d_in[6];    // (2560,4096)
  const float* qnw = (const float*)d_in[7];
  const float* knw = (const float*)d_in[8];
  float* out = (float*)d_out;
  (void)in_sizes;
  (void)n_in;
  (void)out_size;
  (void)ws_size;

  char* ws = (char*)d_ws;
  size_t off = 0;
  auto carve = [&](size_t bytes) {
    char* p = ws + off;
    off += (bytes + 255) & ~(size_t)255;
    return p;
  };
  _Float16* Xh = (_Float16*)carve((size_t)S_LEN * HID * 2);
  _Float16* Wqh = (_Float16*)carve((size_t)DQ * HID * 2);
  _Float16* Wkh = (_Float16*)carve((size_t)DKV * HID * 2);
  _Float16* Wvh = (_Float16*)carve((size_t)DKV * HID * 2);
  _Float16* Woh = (_Float16*)carve((size_t)HID * DQ * 2);
  float* Qf = (float*)carve((size_t)S_LEN * DQ * 4);
  float* Kf = (float*)carve((size_t)S_LEN * DKV * 4);
  float* Vf = (float*)carve((size_t)S_LEN * DKV * 4);
  _Float16* Qh = (_Float16*)carve((size_t)S_LEN * DQ * 2);
  _Float16* Kh = (_Float16*)carve((size_t)S_LEN * DKV * 2);
  _Float16* Vt = (_Float16*)carve((size_t)S_LEN * DKV * 2);
  _Float16* Aoh = (_Float16*)carve((size_t)S_LEN * DQ * 2);

  auto cvt = [&](const float* src, _Float16* dst, size_t n) {
    cvt_f32_f16<<<(unsigned)((n + 255) / 256), 256, 0, stream>>>(src, dst, n);
  };
  cvt(X, Xh, (size_t)S_LEN * HID);
  cvt(Wq, Wqh, (size_t)DQ * HID);
  cvt(Wk, Wkh, (size_t)DKV * HID);
  cvt(Wv, Wvh, (size_t)DKV * HID);
  cvt(Wo, Woh, (size_t)HID * DQ);

  gemm_nt_f16f32<<<dim3(DQ / 64, S_LEN / 64), 128, 0, stream>>>(
      Xh, Wqh, Qf, S_LEN, DQ, HID);
  gemm_nt_f16f32<<<dim3(DKV / 64, S_LEN / 64), 128, 0, stream>>>(
      Xh, Wkh, Kf, S_LEN, DKV, HID);
  gemm_nt_f16f32<<<dim3(DKV / 64, S_LEN / 64), 128, 0, stream>>>(
      Xh, Wvh, Vf, S_LEN, DKV, HID);

  rmsnorm_rope<<<S_LEN * NH / 4, 128, 0, stream>>>(Qf, cosb, sinb, qnw, Qh, NH);
  rmsnorm_rope<<<S_LEN * NKV / 4, 128, 0, stream>>>(Kf, cosb, sinb, knw, Kh,
                                                    NKV);
  v_transpose<<<(S_LEN * DKV + 255) / 256, 256, 0, stream>>>(Vf, Vt);

  attn_fwd<<<NH * (S_LEN / 16) / 4, 128, 0, stream>>>(Qh, Kh, Vt, Aoh);

  gemm_nt_f16f32<<<dim3(HID / 64, S_LEN / 64), 128, 0, stream>>>(
      Aoh, Woh, out, S_LEN, HID, DQ);
}
